// RNNDecoder_34454227648711
// MI455X (gfx1250) — matrix operand order repaired and checked by reading, not compiled, not run on hardware
//
#include <hip/hip_runtime.h>
#include <hip/hip_bf16.h>

typedef __attribute__((ext_vector_type(16))) __bf16 v16bf;
typedef __attribute__((ext_vector_type(8)))  __bf16 v8bf;
typedef __attribute__((ext_vector_type(8)))  float  v8f;
typedef __attribute__((ext_vector_type(4)))  int    v4i;
// address-space-qualified int4 for the async global->LDS DMA builtin
typedef v4i __attribute__((address_space(1))) global_v4i;
typedef v4i __attribute__((address_space(3))) lds_v4i;

namespace {
constexpr int BATCH  = 32;
constexpr int SEQ    = 512;
constexpr int HD     = 512;    // embed = hidden = project
constexpr int G4     = 2048;   // 4*H (gates i,f,c,o)
constexpr int MROW   = BATCH * SEQ; // 16384
constexpr int NLAYER = 2;
}

// Optional CDNA5 async global->LDS DMA path (probe at compile time; the
// fallback keeps the build clean on toolchains without the builtins).
#if defined(__has_builtin)
#if __has_builtin(__builtin_amdgcn_global_load_async_to_lds_b128) && \
    __has_builtin(__builtin_amdgcn_s_wait_asynccnt)
#define USE_ASYNC_LDS 1
#endif
#endif
#ifndef USE_ASYNC_LDS
#define USE_ASYNC_LDS 0
#endif

// ---------------------------------------------------------------------------
// Fragment layouts per CDNA5 ISA 7.12.2 (wave32):
//  A (16x32 bf16): lane = m + half*16 ; half=0 lanes hold K{0..7,16..23},
//                  half=1 lanes hold K{8..15,24..31}; 16 bf16/lane.
//  B (32x16 bf16): lane = n + ((k>>3)&1)*16 ; element e = (k&7) + (k>>4)*8.
//  C/D (16x16 f32): lane = n + (m>=8)*16 ; vgpr v -> m = v + (lane>>4)*8.
// LDS tiles are fragment-major: one fragment read = contiguous 32B per lane.
// ---------------------------------------------------------------------------

__device__ inline v8f wmma_bf16(v16bf a, v16bf b, v8f c) {
  return __builtin_amdgcn_wmma_f32_16x16x32_bf16(false, a, false, b, (short)0, c,
                                                 false, false);
}

__device__ inline v16bf g_loadA(const __bf16* A, int K, int m0, int ar, int akh,
                                int kc) {
  return *(const v16bf*)(A + (size_t)(m0 + ar) * K + kc + akh * 16);
}
__device__ inline void lds_storeA(__bf16* As, int ar, int akh, v16bf v) {
  int mt = ar >> 4, m = ar & 15;
#pragma unroll
  for (int j = 0; j < 16; ++j) {
    int kw = akh * 16 + j;
    int hk = (kw >> 3) & 1;
    int e = (kw & 7) + ((kw >> 4) & 1) * 8;
    As[mt * 512 + (m + hk * 16) * 16 + e] = v[j];
  }
}
__device__ inline v16bf g_loadB(const __bf16* Bm, int N, int n0, int bkw, int bc,
                                int kc) {
  return *(const v16bf*)(Bm + (size_t)(kc + bkw) * N + n0 + bc * 16);
}
__device__ inline void lds_storeB(__bf16* Bs, int bkw, int bc, v16bf v) {
  int lb = ((bkw >> 3) & 1) * 16;
  int e = (bkw & 7) + ((bkw >> 4) & 1) * 8;
#pragma unroll
  for (int j = 0; j < 16; ++j) Bs[bc * 512 + (lb + j) * 16 + e] = v[j];
}

#if USE_ASYNC_LDS
// DMA one A-region (m-tile `w`, k-chunk kc) straight into fragment-major LDS:
// each lane copies two contiguous 16B global chunks to its own 32B LDS slot.
__device__ inline void asyncA(const __bf16* A, int K, int m0, int w, int lane,
                              int kc, __bf16* AsBuf) {
  const __bf16* gp =
      A + (size_t)(m0 + w * 16 + (lane & 15)) * K + kc + (lane >> 4) * 8;
  __bf16* lp = AsBuf + w * 512 + lane * 16;
  __builtin_amdgcn_global_load_async_to_lds_b128((global_v4i*)gp, (lds_v4i*)lp,
                                                 0, 0);
  __builtin_amdgcn_global_load_async_to_lds_b128((global_v4i*)(gp + 16),
                                                 (lds_v4i*)(lp + 8), 0, 0);
}
#endif

// =========================== elementwise kernels ===========================

__global__ __launch_bounds__(256) void k_gather(const int* __restrict__ tokens,
                                                const float* __restrict__ embed,
                                                __bf16* __restrict__ X) {
  int i = blockIdx.x * 256 + threadIdx.x;  // over MROW*HD
  int row = i >> 9;                        // HD == 512
  int e = i & (HD - 1);
  X[i] = (__bf16)embed[(size_t)tokens[row] * HD + e];
}

__global__ __launch_bounds__(256) void k_cvt(const float* __restrict__ s,
                                             __bf16* __restrict__ d, int n) {
  int i = blockIdx.x * 256 + threadIdx.x;
  if (i < n) d[i] = (__bf16)s[i];
}

__global__ __launch_bounds__(256) void k_init(__bf16* __restrict__ h0,
                                              __bf16* __restrict__ h1,
                                              unsigned* __restrict__ bar) {
  int i = blockIdx.x * 256 + threadIdx.x;
  if (i < BATCH * HD) { h0[i] = (__bf16)0.0f; h1[i] = (__bf16)0.0f; }
  if (i == 0) *bar = 0u;
}

// ====================== WMMA GEMM: C = A(bf16) x B(bf16) + bias ============
// Block tile 128x128, 8 waves, K-chunk 32. Double-buffered fragment-major LDS,
// one barrier per chunk, async-DMA (or global->reg) prefetch of chunk k+1
// overlapped with the 8 WMMAs of chunk k. Output mode is a template parameter
// so the epilogue is branch-free.

template <bool WRITE_F, bool WRITE_BF>
__global__ __launch_bounds__(256) void k_gemm_bf16(
    const __bf16* __restrict__ A, const __bf16* __restrict__ Bm,
    const float* __restrict__ bias, float* __restrict__ outF,
    __bf16* __restrict__ outBF, int M, int N, int K) {
  __shared__ __bf16 As[2 * 8 * 512];
  __shared__ __bf16 Bs[2 * 8 * 512];

  const int tid = threadIdx.x;
  const int nb = N >> 7;
  const int bm = blockIdx.x / nb;
  const int bn = blockIdx.x % nb;
  const int m0 = bm << 7, n0 = bn << 7;
  const int w = tid >> 5;   // wave 0..7 (wave32)
  const int wm = w & 3;     // m-tiles {2wm, 2wm+1}
  const int wn = w >> 2;    // n-tiles {4wn .. 4wn+3}
  const int lane = tid & 31;

  v8f acc[2][4];
#pragma unroll
  for (int i = 0; i < 2; ++i)
#pragma unroll
    for (int j = 0; j < 4; ++j) acc[i][j] = {};

  const int ar = tid >> 1, akh = tid & 1;  // A staging: row, k-half
  const int bkw = tid >> 3, bc = tid & 7;  // B staging: k-row, n-tile

  // ---- prologue: start chunk 0 ----
#if USE_ASYNC_LDS
  asyncA(A, K, m0, w, lane, 0, As);
#else
  v16bf apre = g_loadA(A, K, m0, ar, akh, 0);
#endif
  v16bf bpre = g_loadB(Bm, N, n0, bkw, bc, 0);

  int cur = 0;
  for (int kc = 0; kc < K; kc += 32, cur ^= 1) {
    __bf16* Asc = As + cur * (8 * 512);
    __bf16* Bsc = Bs + cur * (8 * 512);
#if USE_ASYNC_LDS
    __builtin_amdgcn_s_wait_asynccnt(0);  // my wave's A-region DMA done
#else
    lds_storeA(Asc, ar, akh, apre);
#endif
    lds_storeB(Bsc, bkw, bc, bpre);
    __syncthreads();  // every wave waited/stored before anyone reads

    if (kc + 32 < K) {  // prefetch next chunk, overlapped with WMMA below
#if USE_ASYNC_LDS
      asyncA(A, K, m0, w, lane, kc + 32, As + (cur ^ 1) * (8 * 512));
#else
      apre = g_loadA(A, K, m0, ar, akh, kc + 32);
#endif
      bpre = g_loadB(Bm, N, n0, bkw, bc, kc + 32);
    }

    v16bf af[2], bfr[4];
#pragma unroll
    for (int i = 0; i < 2; ++i)
      af[i] = *(const v16bf*)(Asc + (wm * 2 + i) * 512 + lane * 16);
#pragma unroll
    for (int j = 0; j < 4; ++j)
      bfr[j] = *(const v16bf*)(Bsc + (wn * 4 + j) * 512 + lane * 16);
#pragma unroll
    for (int i = 0; i < 2; ++i)
#pragma unroll
      for (int j = 0; j < 4; ++j)
        acc[i][j] = wmma_bf16(af[i], bfr[j], acc[i][j]);
  }

  // ---- branch-free epilogue (C/D layout -> global) ----
  const int nn = lane & 15, rb = (lane >> 4) * 8;
#pragma unroll
  for (int i = 0; i < 2; ++i) {
#pragma unroll
    for (int j = 0; j < 4; ++j) {
      int gm0 = m0 + (wm * 2 + i) * 16;
      int gn = n0 + (wn * 4 + j) * 16 + nn;
      float bi = bias[gn];
#pragma unroll
      for (int v = 0; v < 8; ++v) {
        float val = acc[i][j][v] + bi;
        size_t o = (size_t)(gm0 + rb + v) * N + gn;
        if (WRITE_F) outF[o] = val;
        if (WRITE_BF) outBF[o] = (__bf16)val;
      }
    }
  }
}

// ========================= LSTM recurrence (scan) ==========================
// Grid = 16 workgroups (one cluster's worth, one per WGP), 256 threads.
// Workgroup wg owns hidden columns [wg*32, wg*32+32): its 512x128 bf16 slice
// of U lives in LDS (128 KB of the 320 KB WGP pool) for all 512 timesteps.
// Per step: [32x512]x[512x128] via WMMA, gate math, h broadcast through L2,
// cluster barrier (S_NOP when not cluster-dispatched) + atomic fallback.
// h ping-pong pointers are loop-carried (PHI of AS1 pointers) so the h_prev
// fragment loads lower as global_load_b128, not flat_load (which would tie up
// DScnt/LDS bandwidth in the latency-critical chain).

__global__ __launch_bounds__(256) void k_lstm_scan(
    const float* __restrict__ XG, const __bf16* __restrict__ Ubf,
    __bf16* __restrict__ h0, __bf16* __restrict__ h1, float* __restrict__ HS,
    unsigned* __restrict__ bar, int nblk) {
  extern __shared__ char smem[];
  __bf16* Us = (__bf16*)smem;                     // 16 kchunks * 8 nt * 512
  float* gb = (float*)(smem + 16 * 8 * 512 * 2);  // 32x128 partial gates

  const int tid = threadIdx.x;
  const int wg = blockIdx.x;
  const int w = tid >> 5, lane = tid & 31;
  const int mt = w & 1;   // row tile (B=32 -> 2 tiles of 16)
  const int ng = w >> 1;  // n-tile pair {2ng, 2ng+1}

  // ---- stage U slice into fragment-major LDS (once, re-read 512x) ----
  for (int idx = tid; idx < HD * 128; idx += 256) {
    int kg = idx >> 7, nl = idx & 127;
    int gate = nl >> 5, c = nl & 31;
    __bf16 v = Ubf[(size_t)kg * G4 + gate * HD + wg * 32 + c];
    int kcq = kg >> 5, kw = kg & 31;
    int nt = nl >> 4, n = nl & 15;
    Us[(kcq * 8 + nt) * 512 + (n + ((kw >> 3) & 1) * 16) * 16 + (kw & 7) + ((kw >> 4) & 1) * 8] = v;
  }
  __syncthreads();

  float creg[4] = {0.f, 0.f, 0.f, 0.f};  // cell state, 4 cells per thread
  __bf16* hin = h0;   // t=0 reads h0 (zeroed by k_init)
  __bf16* hout = h1;

  for (int t = 0; t < SEQ; ++t) {
    v8f a0 = {}, a1 = {};
#pragma unroll 4
    for (int kc = 0; kc < 16; ++kc) {
      // A fragment of h_prev straight from (L2-hot) global
      const __bf16* p =
          hin + (size_t)(mt * 16 + (lane & 15)) * HD + kc * 32 + (lane >> 4) * 8;
      v8bf lo = *(const v8bf*)p;
      v8bf hi = *(const v8bf*)(p + 16);
      v16bf a;
#pragma unroll
      for (int e = 0; e < 8; ++e) { a[e] = lo[e]; a[8 + e] = hi[e]; }
      v16bf b0 = *(const v16bf*)(Us + (kc * 8 + ng * 2 + 0) * 512 + lane * 16);
      v16bf b1 = *(const v16bf*)(Us + (kc * 8 + ng * 2 + 1) * 512 + lane * 16);
      a0 = wmma_bf16(a, b0, a0);
      a1 = wmma_bf16(a, b1, a1);
    }
    // partial pre-activations -> LDS (cols are gate*32 + c)
    {
      int n = lane & 15, rb = (lane >> 4) * 8;
#pragma unroll
      for (int v = 0; v < 8; ++v) {
        gb[(mt * 16 + rb + v) * 128 + ng * 32 + n] = a0[v];
        gb[(mt * 16 + rb + v) * 128 + ng * 32 + 16 + n] = a1[v];
      }
    }
    __syncthreads();

    // prefetch next timestep's XG row (global_prefetch_b8)
    if (t + 1 < SEQ) {
      int bb0 = tid >> 5, c0 = tid & 31;
      const float* nx = XG + (size_t)(bb0 * SEQ + t + 1) * G4 + wg * 32 + c0;
      __builtin_prefetch(nx, 0, 1);
      __builtin_prefetch(nx + HD, 0, 1);
      __builtin_prefetch(nx + 2 * HD, 0, 1);
      __builtin_prefetch(nx + 3 * HD, 0, 1);
    }

    // gate math for this wg's 32x32 (batch x hidden-col) cells
#pragma unroll
    for (int k2 = 0; k2 < 4; ++k2) {
      int idx = k2 * 256 + tid;
      int bb = idx >> 5, c = idx & 31;
      int hc = wg * 32 + c;
      const float* xg = XG + (size_t)(bb * SEQ + t) * G4;
      float gi = xg[hc] + gb[bb * 128 + c];
      float gf = xg[HD + hc] + gb[bb * 128 + 32 + c];
      float gc = xg[2 * HD + hc] + gb[bb * 128 + 64 + c];
      float go = xg[3 * HD + hc] + gb[bb * 128 + 96 + c];
      gi = 1.0f / (1.0f + __expf(-gi));
      gf = 1.0f / (1.0f + __expf(-gf));
      go = 1.0f / (1.0f + __expf(-go));
      float cn = gf * creg[k2] + gi * tanhf(gc);
      creg[k2] = cn;
      float hv = go * tanhf(cn);
      hout[(size_t)bb * HD + hc] = (__bf16)hv;
      HS[(size_t)(bb * SEQ + t) * HD + hc] = hv;
    }
    __syncthreads();
    __threadfence();                       // h visible at device scope
    __builtin_amdgcn_s_cluster_barrier();  // fast path under cluster dispatch
    if (tid == 0) {                        // portable fallback grid barrier
      unsigned vv = atomicAdd(bar, 1u) + 1u;
      unsigned tgt =
          ((vv + (unsigned)nblk - 1u) / (unsigned)nblk) * (unsigned)nblk;
      while (__hip_atomic_load(bar, __ATOMIC_ACQUIRE, __HIP_MEMORY_SCOPE_AGENT) <
             tgt)
        __builtin_amdgcn_s_sleep(1);
    }
    __syncthreads();
    __threadfence();

    __bf16* tmp = hin;  // ping-pong (loop-carried PHI keeps AS1 inference)
    hin = hout;
    hout = tmp;
  }
}

// ============================== LayerNorm ==================================

__global__ __launch_bounds__(256) void k_layernorm(const float* __restrict__ X,
                                                   const float* __restrict__ g,
                                                   const float* __restrict__ be,
                                                   __bf16* __restrict__ out) {
  __shared__ float red[256];
  const int row = blockIdx.x;
  const float* x = X + (size_t)row * HD;
  float s = 0.f;
  for (int j = threadIdx.x; j < HD; j += 256) s += x[j];
  red[threadIdx.x] = s;
  __syncthreads();
  for (int st = 128; st > 0; st >>= 1) {
    if (threadIdx.x < st) red[threadIdx.x] += red[threadIdx.x + st];
    __syncthreads();
  }
  float mu = red[0] / (float)HD;
  __syncthreads();
  float v = 0.f;
  for (int j = threadIdx.x; j < HD; j += 256) {
    float d = x[j] - mu;
    v += d * d;
  }
  red[threadIdx.x] = v;
  __syncthreads();
  for (int st = 128; st > 0; st >>= 1) {
    if (threadIdx.x < st) red[threadIdx.x] += red[threadIdx.x + st];
    __syncthreads();
  }
  float rs = rsqrtf(red[0] / (float)HD + 1e-3f);
  for (int j = threadIdx.x; j < HD; j += 256)
    out[(size_t)row * HD + j] = (__bf16)((x[j] - mu) * rs * g[j] + be[j]);
}

// ============================== launcher ===================================

extern "C" void kernel_launch(void* const* d_in, const int* in_sizes, int n_in,
                              void* d_out, int out_size, void* d_ws,
                              size_t ws_size, hipStream_t stream) {
  (void)in_sizes; (void)n_in; (void)out_size; (void)ws_size;
  const int* tokens = (const int*)d_in[0];
  const float* embed = (const float*)d_in[1];
  const float* W = (const float*)d_in[2];
  const float* U = (const float*)d_in[3];
  const float* bvec = (const float*)d_in[4];
  const float* gamma = (const float*)d_in[5];
  const float* beta = (const float*)d_in[6];
  const float* Wd = (const float*)d_in[7];
  const float* bd = (const float*)d_in[8];

  size_t off = 0;
  auto give = [&](size_t bytes) -> char* {
    char* p = (char*)d_ws + off;
    off = (off + bytes + 255) & ~(size_t)255;
    return p;
  };
  __bf16* bufX = (__bf16*)give((size_t)MROW * HD * 2);
  __bf16* bufXLN = (__bf16*)give((size_t)MROW * HD * 2);
  __bf16* Wbf = (__bf16*)give((size_t)NLAYER * HD * G4 * 2);
  __bf16* Ubf = (__bf16*)give((size_t)NLAYER * HD * G4 * 2);
  __bf16* Wdbf = (__bf16*)give((size_t)NLAYER * HD * HD * 2);
  float* XG = (float*)give((size_t)MROW * G4 * 4);
  float* HS = (float*)give((size_t)MROW * HD * 4);
  __bf16* h0 = (__bf16*)give((size_t)BATCH * HD * 2);
  __bf16* h1 = (__bf16*)give((size_t)BATCH * HD * 2);
  unsigned* bar = (unsigned*)give(256);

  // embedding gather (f32 -> bf16) + weight conversion
  k_gather<<<MROW * HD / 256, 256, 0, stream>>>(tokens, embed, bufX);
  k_cvt<<<(NLAYER * HD * G4 + 255) / 256, 256, 0, stream>>>(W, Wbf,
                                                            NLAYER * HD * G4);
  k_cvt<<<(NLAYER * HD * G4 + 255) / 256, 256, 0, stream>>>(U, Ubf,
                                                            NLAYER * HD * G4);
  k_cvt<<<(NLAYER * HD * HD + 255) / 256, 256, 0, stream>>>(Wd, Wdbf,
                                                            NLAYER * HD * HD);

  const size_t scan_lds = (size_t)16 * 8 * 512 * 2 + (size_t)32 * 128 * 4;

  for (int l = 0; l < NLAYER; ++l) {
    // XG = X @ W_l + b_l   [16384 x 2048], f32 out
    k_gemm_bf16<true, false><<<(MROW / 128) * (G4 / 128), 256, 0, stream>>>(
        bufX, Wbf + (size_t)l * HD * G4, bvec + l * G4, XG, nullptr, MROW, G4,
        HD);
    // recurrence
    k_init<<<(BATCH * HD + 255) / 256, 256, 0, stream>>>(h0, h1, bar);
    k_lstm_scan<<<16, 256, scan_lds, stream>>>(XG, Ubf + (size_t)l * HD * G4,
                                               h0, h1, HS, bar, 16);
    // LayerNorm (before Dense, matching reference) then Dense
    k_layernorm<<<MROW, 256, 0, stream>>>(HS, gamma + l * HD, beta + l * HD,
                                          bufXLN);
    if (l == NLAYER - 1) {
      k_gemm_bf16<true, false><<<(MROW / 128) * (HD / 128), 256, 0, stream>>>(
          bufXLN, Wdbf + (size_t)l * HD * HD, bd + l * HD, (float*)d_out,
          nullptr, MROW, HD, HD);
    } else {
      k_gemm_bf16<false, true><<<(MROW / 128) * (HD / 128), 256, 0, stream>>>(
          bufXLN, Wdbf + (size_t)l * HD * HD, bd + l * HD, nullptr, bufX, MROW,
          HD, HD);
    }
  }
}
